// ActorCritic_55155970015325
// MI455X (gfx1250) — matrix operand order repaired and checked by reading, not compiled, not run on hardware
//
#include <hip/hip_runtime.h>
#include <cstdint>

typedef __attribute__((ext_vector_type(16))) _Float16 v16h;
typedef __attribute__((ext_vector_type(8)))  _Float16 v8h;
typedef __attribute__((ext_vector_type(8)))  float    v8f;
typedef unsigned int u32x4 __attribute__((ext_vector_type(4)));
typedef int i32x4 __attribute__((ext_vector_type(4)));
typedef int i32x8 __attribute__((ext_vector_type(8)));
typedef _Float16 half_t;

typedef __attribute__((address_space(1))) int gint_t;
typedef __attribute__((address_space(3))) int lint_t;

// Generic->addrspace pointer conversion via inttoptr.
// For LDS, the low 32 bits of the generic (aperture-based) address are the
// LDS offset (ISA 10.2).
static __device__ __forceinline__ gint_t* as_global_i(const void* p) {
  return (gint_t*)(unsigned long long)(uintptr_t)p;
}
static __device__ __forceinline__ lint_t* as_lds_i(const void* p) {
  return (lint_t*)(unsigned int)(uintptr_t)p;
}

#define B_SAMPLES 4096
#define ZDIM 256
#define HDIM 1024
#define OPT_STRIDE 22688
#define OFF_B1 16384
#define OFF_W2 16448
#define OFF_B2 20544
#define OFF_W3 20608
#define OFF_B3 22656
#define OUT_STRIDE 33

// ---------------------------------------------------------------- converts
__global__ __launch_bounds__(256) void k_f32_to_f16(const float* __restrict__ src,
                                                    half_t* __restrict__ dst, int n) {
  int i = blockIdx.x * 256 + threadIdx.x;
  if (i < n) dst[i] = (half_t)src[i];
}

__device__ __forceinline__ float elu_f(float x) {
  return x > 0.f ? x : (__expf(x) - 1.f);
}

// ---------------------------------------------------------------- WMMA GEMM
// Out[M,N] = elu(A[M,K] @ W[N,K]^T + bias[N]), f16 in/out, f32 accumulate.
// One wave computes one 16x16 output tile via v_wmma_f32_16x16x32_f16.
__global__ __launch_bounds__(128) void k_gemm_elu_f16(
    const half_t* __restrict__ A,     // [M,K] row-major
    const half_t* __restrict__ W,     // [N,K] row-major
    const float*  __restrict__ bias,  // [N]
    half_t* __restrict__ Out,         // [M,N] row-major
    int M, int N, int K) {
  const int wave = threadIdx.x >> 5;
  const int lane = threadIdx.x & 31;
  const int row0 = blockIdx.x * 16;
  const int col0 = (blockIdx.y * 4 + wave) * 16;
  const int g  = lane >> 4;
  const int id = lane & 15;

  const half_t* __restrict__ arow = A + (size_t)(row0 + id) * K;
  const half_t* __restrict__ wcol = W + (size_t)(col0 + id) * K;

  v8f c = {};
  for (int k0 = 0; k0 < K; k0 += 32) {
    union { v16h v; v8h h[2]; } a, b;
    a.h[0] = *(const v8h*)(arow + k0 + g * 16);
    a.h[1] = *(const v8h*)(arow + k0 + g * 16 + 8);
    b.v    = *(const v16h*)(wcol + k0 + g * 16);
    c = __builtin_amdgcn_wmma_f32_16x16x32_f16(
        false, a.v, false, b.v, (short)0, c, false, false);
  }

  const int n = col0 + id;
  const float bn = bias[n];
#pragma unroll
  for (int r = 0; r < 8; ++r) {
    const int m = row0 + r + g * 8;
    float v = c[r] + bn;
    v = elu_f(v);
    Out[(size_t)m * N + n] = (half_t)v;
  }
}

// ---------------------------------------------------------------- value dot
__global__ __launch_bounds__(64) void k_value_out(const half_t* __restrict__ T2,
                                                  const float* __restrict__ w3,
                                                  const float* __restrict__ b3,
                                                  float* __restrict__ out) {
  const int b = blockIdx.x * 2 + (threadIdx.x >> 5);
  const int lane = threadIdx.x & 31;
  const half_t* __restrict__ row = T2 + (size_t)b * HDIM;
  float acc = 0.f;
#pragma unroll 8
  for (int i = lane; i < HDIM; i += 32)
    acc = fmaf((float)row[i], w3[i], acc);
#pragma unroll
  for (int off = 16; off > 0; off >>= 1)
    acc += __shfl_xor(acc, off, 32);
  if (lane == 0) out[(size_t)b * OUT_STRIDE] = acc + b3[0];
}

// ---------------------------------------------------------------- hypernet
// One 256-thread workgroup per sample. The 88.6KB per-sample parameter blob
// is staged into LDS by the Tensor Data Mover (one DMA per workgroup,
// TENSORcnt-tracked); the 1KB input vector via async global->LDS copies
// (ASYNCcnt-tracked). Compute is conflict-free rotated LDS dot products.
__global__ __launch_bounds__(256) void k_hyper(const float* __restrict__ inputs,
                                               const float* __restrict__ option,
                                               float* __restrict__ out) {
  __shared__ float s_opt[OPT_STRIDE];  // 90752 B
  __shared__ float s_z[ZDIM];
  __shared__ float s_h1[64];
  __shared__ float s_h2[64];

  const int b = blockIdx.x;
  const int t = threadIdx.x;

  // ---- stage z[256] ----
#if __has_builtin(__builtin_amdgcn_global_load_async_to_lds_b32) && \
    __has_builtin(__builtin_amdgcn_s_wait_asynccnt)
  __builtin_amdgcn_global_load_async_to_lds_b32(
      as_global_i(inputs + (size_t)b * ZDIM + t), as_lds_i(&s_z[t]), 0, 0);
#else
  s_z[t] = inputs[(size_t)b * ZDIM + t];
#endif

  // ---- stage option blob (88.6KB) ----
#if __has_builtin(__builtin_amdgcn_tensor_load_to_lds) && \
    __has_builtin(__builtin_amdgcn_s_wait_tensorcnt)
  if (t < 32) {  // wave 0 issues one TDM descriptor for the whole blob
    const unsigned long long ga =
        (unsigned long long)(uintptr_t)(option + (size_t)b * OPT_STRIDE);
    const unsigned int lds_off = (unsigned int)(uintptr_t)&s_opt[0];
    // D# group 0: count=1 | lds_addr | global_addr[56:0] | type=2
    u32x4 g0 = {1u, lds_off, (unsigned int)ga,
                (unsigned int)(((ga >> 32) & 0x01FFFFFFull) | (2u << 30))};
    // D# group 1: data_size=4B; tensor_dim0=22688, tensor_dim1=1;
    // tile_dim0=22688, tile_dim1=1; tensor_dim0_stride=22688.
    i32x8 g1 = {(int)(2u << 16),
                (int)((OPT_STRIDE & 0xFFFF) << 16),
                (int)(((OPT_STRIDE >> 16) & 0xFFFF) | (1 << 16)),
                (int)((OPT_STRIDE & 0xFFFF) << 16),
                1,
                OPT_STRIDE,
                0, 0};
    i32x4 gz = {0, 0, 0, 0};
#if __clang_major__ >= 23
    i32x8 gz8 = {0, 0, 0, 0, 0, 0, 0, 0};
    __builtin_amdgcn_tensor_load_to_lds(g0, g1, gz, gz, gz8, 0);
#else
    __builtin_amdgcn_tensor_load_to_lds(g0, g1, gz, gz, 0);
#endif
    __builtin_amdgcn_s_wait_tensorcnt(0);
  }
#else
  {
    const float4* __restrict__ o4 = (const float4*)(option + (size_t)b * OPT_STRIDE);
    float4* s4 = (float4*)s_opt;
    for (int i = t; i < OPT_STRIDE / 4; i += 256) s4[i] = o4[i];
  }
#endif

#if __has_builtin(__builtin_amdgcn_global_load_async_to_lds_b32) && \
    __has_builtin(__builtin_amdgcn_s_wait_asynccnt)
  __builtin_amdgcn_s_wait_asynccnt(0);
#endif
  __syncthreads();

  // ---- layer 1: h1 = relu(W1[64,256] @ z + b1) ; 4 lanes per row ----
  {
    const int r = t >> 2, q = t & 3;
    const float* __restrict__ wrow = s_opt + r * 256 + q * 64;
    const float* __restrict__ zq = s_z + q * 64;
    const int rot = t & 63;  // lane rotation -> 32 distinct LDS banks
    float acc = 0.f;
#pragma unroll 8
    for (int i = 0; i < 64; ++i) {
      const int j = (i + rot) & 63;
      acc = fmaf(wrow[j], zq[j], acc);
    }
    acc += __shfl_xor(acc, 1, 32);
    acc += __shfl_xor(acc, 2, 32);
    if (q == 0) s_h1[r] = fmaxf(acc + s_opt[OFF_B1 + r], 0.f);
  }
  __syncthreads();

  // ---- layer 2: h2 = relu(W2[64,64] @ h1 + b2) ; 4 lanes per row ----
  {
    const int r = t >> 2, q = t & 3;
    const float* __restrict__ wrow = s_opt + OFF_W2 + r * 64 + q * 16;
    const float* __restrict__ hq = s_h1 + q * 16;
    const int rot = (t + (t >> 2)) & 15;  // injective (q,j) map -> no conflicts
    float acc = 0.f;
#pragma unroll
    for (int i = 0; i < 16; ++i) {
      const int j = (i + rot) & 15;
      acc = fmaf(wrow[j], hq[j], acc);
    }
    acc += __shfl_xor(acc, 1, 32);
    acc += __shfl_xor(acc, 2, 32);
    if (q == 0) s_h2[r] = fmaxf(acc + s_opt[OFF_B2 + r], 0.f);
  }
  __syncthreads();

  // ---- layer 3: a = W3[32,64] @ h2 + b3 ; 8 lanes per row ----
  {
    const int r = t >> 3, q = t & 7;
    const float* __restrict__ wrow = s_opt + OFF_W3 + r * 64 + q * 8;
    const float* __restrict__ hq = s_h2 + q * 8;
    const int rot = (t + (t >> 3)) & 7;
    float acc = 0.f;
#pragma unroll
    for (int i = 0; i < 8; ++i) {
      const int j = (i + rot) & 7;
      acc = fmaf(wrow[j], hq[j], acc);
    }
    acc += __shfl_xor(acc, 1, 32);
    acc += __shfl_xor(acc, 2, 32);
    acc += __shfl_xor(acc, 4, 32);
    if (q == 0) out[(size_t)b * OUT_STRIDE + 1 + r] = acc + s_opt[OFF_B3 + r];
  }
}

// ---------------------------------------------------------------- launcher
extern "C" void kernel_launch(void* const* d_in, const int* in_sizes, int n_in,
                              void* d_out, int out_size, void* d_ws, size_t ws_size,
                              hipStream_t stream) {
  (void)in_sizes; (void)n_in; (void)out_size; (void)ws_size;
  const float* inputs = (const float*)d_in[0];
  const float* option = (const float*)d_in[1];
  const float* w1 = (const float*)d_in[2];
  const float* b1 = (const float*)d_in[3];
  const float* w2 = (const float*)d_in[4];
  const float* b2 = (const float*)d_in[5];
  const float* w3 = (const float*)d_in[6];
  const float* b3 = (const float*)d_in[7];
  float* out = (float*)d_out;

  // workspace layout (f16): Xh | W1h | W2h | T1h | T2h  (~20.5 MB total)
  half_t* Xh  = (half_t*)d_ws;
  half_t* W1h = Xh + (size_t)B_SAMPLES * ZDIM;
  half_t* W2h = W1h + (size_t)HDIM * ZDIM;
  half_t* T1h = W2h + (size_t)HDIM * HDIM;
  half_t* T2h = T1h + (size_t)B_SAMPLES * HDIM;

  const int nX = B_SAMPLES * ZDIM;
  k_f32_to_f16<<<(nX + 255) / 256, 256, 0, stream>>>(inputs, Xh, nX);
  const int nW1 = HDIM * ZDIM;
  k_f32_to_f16<<<(nW1 + 255) / 256, 256, 0, stream>>>(w1, W1h, nW1);
  const int nW2 = HDIM * HDIM;
  k_f32_to_f16<<<(nW2 + 255) / 256, 256, 0, stream>>>(w2, W2h, nW2);

  // hypernet first: it is the HBM-bound stream; overlap WMMA GEMMs behind it.
  k_hyper<<<B_SAMPLES, 256, 0, stream>>>(inputs, option, out);

  dim3 gg(B_SAMPLES / 16, HDIM / 64);  // 16x16 tile per wave, 4 waves per block
  k_gemm_elu_f16<<<gg, 128, 0, stream>>>(Xh, W1h, b1, T1h, B_SAMPLES, HDIM, ZDIM);
  k_gemm_elu_f16<<<gg, 128, 0, stream>>>(T1h, W2h, b2, T2h, B_SAMPLES, HDIM, HDIM);

  k_value_out<<<B_SAMPLES / 2, 64, 0, stream>>>(T2h, w3, b3, out);
}
